// RNAGraphTransformer_12730283065541
// MI455X (gfx1250) — matrix operand order repaired and checked by reading, not compiled, not run on hardware
//
#include <hip/hip_runtime.h>
#include <math.h>

typedef __attribute__((ext_vector_type(16))) __bf16 v16bf;
typedef __attribute__((ext_vector_type(8)))  float  v8f;

union BF16Frag { v16bf v; uint4 q[2]; };

__device__ __forceinline__ __bf16 f2bf(float f) {
    unsigned u = __float_as_uint(f);
    u += 0x7FFFu + ((u >> 16) & 1u);              // round-to-nearest-even
    unsigned short s = (unsigned short)(u >> 16);
    return __builtin_bit_cast(__bf16, s);
}
__device__ __forceinline__ float bf2f(__bf16 x) {
    unsigned short s = __builtin_bit_cast(unsigned short, x);
    return __uint_as_float(((unsigned)s) << 16);
}
__device__ __forceinline__ float gelu_exact(float x) {
    return 0.5f * x * (1.0f + erff(x * 0.70710678118654752f));
}
// sum over a 128-thread block (4 wave32)
__device__ __forceinline__ float blocksum128(float v, float* sh) {
    for (int off = 16; off > 0; off >>= 1) v += __shfl_xor(v, off, 32);
    __syncthreads();
    if ((threadIdx.x & 31) == 0) sh[threadIdx.x >> 5] = v;
    __syncthreads();
    return sh[0] + sh[1] + sh[2] + sh[3];
}

// ---------------- weight packing (f32 [K][OUT] -> bf16 [OUT][K]) ----------------
__global__ void conv_t_bf16(const float* __restrict__ src, __bf16* __restrict__ dst,
                            int IN, int OUT) {
    int idx = blockIdx.x * blockDim.x + threadIdx.x;
    if (idx >= IN * OUT) return;
    int o = idx / IN, i = idx - o * IN;
    dst[idx] = f2bf(src[i * OUT + o]);
}

// pack q,k,v,skip weights for one layer into [512][128] bf16 + fused bias[512]
__global__ void pack_qkvs(const float* __restrict__ wq, const float* __restrict__ wk,
                          const float* __restrict__ wv, const float* __restrict__ ws,
                          const float* __restrict__ bq, const float* __restrict__ bk,
                          const float* __restrict__ bv, const float* __restrict__ bs,
                          __bf16* __restrict__ wt, float* __restrict__ bias) {
    int idx = blockIdx.x * blockDim.x + threadIdx.x;
    if (idx >= 512 * 128) return;
    int o = idx >> 7, i = idx & 127;
    int seg = o >> 7, oo = o & 127;
    const float* w = (seg == 0) ? wq : (seg == 1) ? wk : (seg == 2) ? wv : ws;
    wt[idx] = f2bf(w[i * 128 + oo]);
    if (i == 0) {
        const float* bb = (seg == 0) ? bq : (seg == 1) ? bk : (seg == 2) ? bv : bs;
        bias[o] = bb[oo];
    }
}

// ---------------- WMMA GEMM: C[N x OUT] = A_bf16[N x K] * Bt_bf16[OUT x K] + bias ----
// blockDim.x = 32*(OUT/64); wave w computes a 16x64 tile (4 accumulators, A reused).
// Double-buffered fragments: loads for k-step s+1 issue before the WMMAs of step s.
template <int KDIM>
__global__ void wmma_gemm_bf16(const __bf16* __restrict__ A, const __bf16* __restrict__ Bt,
                               const float* __restrict__ bias, float* __restrict__ C,
                               int OUT) {
    constexpr int STEPS = KDIM / 32;
    const int lane = threadIdx.x & 31;
    const int wave = threadIdx.x >> 5;
    const int m    = lane & 15;
    const int half = lane >> 4;
    const long row = (long)blockIdx.x * 16 + m;    // A row for this lane
    const int col0 = wave * 64;
    const __bf16* arow = A + row * KDIM;
    const __bf16* brow[4];
    #pragma unroll
    for (int j = 0; j < 4; j++)
        brow[j] = Bt + (long)(col0 + j * 16 + m) * KDIM + half * 8;

    v8f acc[4] = {{}, {}, {}, {}};
    BF16Frag a[2];
    BF16Frag b[2][4];

    // preload k-step 0
    // A 16x32 fragment: lanes 0-15 hold K[0..7],[16..23]; lanes 16-31 K[8..15],[24..31]
    a[0].q[0] = *(const uint4*)(arow + half * 8);
    a[0].q[1] = *(const uint4*)(arow + 16 + half * 8);
    #pragma unroll
    for (int j = 0; j < 4; j++) {
        // B 32x16 fragment: lanes 0-15 hold K[0..7],[16..23]; lanes 16-31 K[8..15],[24..31] of column n
        b[0][j].q[0] = *(const uint4*)(brow[j]);
        b[0][j].q[1] = *(const uint4*)(brow[j] + 16);
    }

    #pragma unroll
    for (int s = 0; s < STEPS; s++) {
        const int cur = s & 1, nxt = cur ^ 1;
        if (s + 1 < STEPS) {
            const int kk = (s + 1) * 32;
            a[nxt].q[0] = *(const uint4*)(arow + kk + half * 8);
            a[nxt].q[1] = *(const uint4*)(arow + kk + 16 + half * 8);
            #pragma unroll
            for (int j = 0; j < 4; j++) {
                b[nxt][j].q[0] = *(const uint4*)(brow[j] + kk);
                b[nxt][j].q[1] = *(const uint4*)(brow[j] + kk + 16);
            }
        }
        #pragma unroll
        for (int j = 0; j < 4; j++)
            acc[j] = __builtin_amdgcn_wmma_f32_16x16x32_bf16(false, a[cur].v, false, b[cur][j].v,
                                                             (short)0, acc[j], false, false);
    }

    const long r0 = (long)blockIdx.x * 16 + half * 8;
    #pragma unroll
    for (int j = 0; j < 4; j++) {
        const int col = col0 + j * 16 + m;
        const float bn = bias[col];
        #pragma unroll
        for (int r = 0; r < 8; r++)
            C[(r0 + r) * OUT + col] = acc[j][r] + bn;
    }
}

// ---------------- encoder: h = gelu(ln(x@W+b)) ----------------
__global__ void encoder_kernel(const float* __restrict__ x, const float* __restrict__ w,
                               const float* __restrict__ b, const float* __restrict__ g,
                               const float* __restrict__ bn, float* __restrict__ h,
                               __bf16* __restrict__ hb) {
    __shared__ float xs[20];
    __shared__ float sh[4];
    const int n = blockIdx.x, c = threadIdx.x;
    if (c < 20) xs[c] = x[n * 20 + c];
    __syncthreads();
    float s = b[c];
    #pragma unroll
    for (int k = 0; k < 20; k++) s += xs[k] * w[k * 128 + c];
    float mean = blocksum128(s, sh) * (1.0f / 128.0f);
    float d = s - mean;
    float var = blocksum128(d * d, sh) * (1.0f / 128.0f);
    float y = d * rsqrtf(var + 1e-5f) * g[c] + bn[c];
    y = gelu_exact(y);
    h[(long)n * 128 + c]  = y;
    hb[(long)n * 128 + c] = f2bf(y);
}

// ---------------- CSR build (done once; edge_index constant across layers) ----------
__global__ void deg_kernel(const int* __restrict__ ei, int* __restrict__ deg, int E) {
    int e = blockIdx.x * blockDim.x + threadIdx.x;
    if (e < E) atomicAdd(&deg[ei[E + e]], 1);
}

// single-block hierarchical exclusive scan (1024 threads = 32 waves)
__global__ void scan_kernel(const int* __restrict__ deg, int* __restrict__ rowstart, int N) {
    __shared__ int sh[32];
    const int t = threadIdx.x, lane = t & 31, w = t >> 5;
    int carry = 0;
    for (int base = 0; base < N; base += 1024) {
        int i = base + t;
        int v = (i < N) ? deg[i] : 0;
        int x = v;
        for (int off = 1; off < 32; off <<= 1) {
            int y = __shfl_up(x, off, 32);
            if (lane >= off) x += y;
        }
        if (lane == 31) sh[w] = x;
        __syncthreads();
        if (w == 0) {
            int tt = sh[lane];
            for (int off = 1; off < 32; off <<= 1) {
                int y = __shfl_up(tt, off, 32);
                if (lane >= off) tt += y;
            }
            sh[lane] = tt;
        }
        __syncthreads();
        int waveoff = (w > 0) ? sh[w - 1] : 0;
        if (i < N) rowstart[i] = x + waveoff + carry - v;   // exclusive
        int blocktotal = sh[31];
        __syncthreads();
        carry += blocktotal;
    }
    if (t == 0) rowstart[N] = carry;
}

__global__ void fill_kernel(const int* __restrict__ ei, const int* __restrict__ rowstart,
                            int* __restrict__ cursor, int* __restrict__ elist, int E) {
    int e = blockIdx.x * blockDim.x + threadIdx.x;
    if (e >= E) return;
    int dst = ei[E + e];
    int pos = atomicAdd(&cursor[dst], 1);
    elist[rowstart[dst] + pos] = e;
}

// ---------------- edge pass: raw attention logits (no atomics) ----------------
__global__ void edge_alpha(const int* __restrict__ ei, const float* __restrict__ eattr,
                           const float* __restrict__ qkvs, const float* __restrict__ wel,
                           const float* __restrict__ bel, float* __restrict__ alpha, int E) {
    const int lane = threadIdx.x & 31;
    const long e = (long)blockIdx.x * 8 + (threadIdx.x >> 5);
    if (e >= E) return;
    const int src = ei[e], dst = ei[E + e];
    const float ea = eattr[e];
    const float* q = qkvs + (long)dst * 512;
    const float* k = qkvs + (long)src * 512 + 128;
    float p[4];
    #pragma unroll
    for (int hh = 0; hh < 4; hh++) {
        int c = hh * 32 + lane;
        float ef = ea * wel[c] + bel[c];
        p[hh] = q[c] * (k[c] + ef);
    }
    for (int off = 16; off > 0; off >>= 1) {
        #pragma unroll
        for (int hh = 0; hh < 4; hh++) p[hh] += __shfl_xor(p[hh], off, 32);
    }
    if (lane == 0) {
        #pragma unroll
        for (int hh = 0; hh < 4; hh++)
            alpha[e * 4 + hh] = p[hh] * 0.17677669529663687f;   // 1/sqrt(32)
    }
}

// ---------------- per-node softmax + gather-aggregate (no atomics) ----------------
__global__ void node_softmax_agg(const int* __restrict__ ei, const float* __restrict__ eattr,
                                 const int* __restrict__ rowstart, const int* __restrict__ elist,
                                 const float* __restrict__ qkvs, const float* __restrict__ wel,
                                 const float* __restrict__ bel, const float* __restrict__ alpha,
                                 float* __restrict__ agg, int E) {
    __shared__ int   s_src[128];
    __shared__ float s_ea[128];
    __shared__ float s_a[4][128];
    __shared__ float s_w[4][4];
    __shared__ float s_amax[4], s_den[4];
    const int n = blockIdx.x, t = threadIdx.x;
    const int lane = t & 31, w = t >> 5;
    const int start = rowstart[n];
    const int deg   = rowstart[n + 1] - start;

    // pass 1: per-head max over in-edges
    float mx[4] = {-1e30f, -1e30f, -1e30f, -1e30f};
    for (int j = t; j < deg; j += 128) {
        long e = elist[start + j];
        #pragma unroll
        for (int hh = 0; hh < 4; hh++) mx[hh] = fmaxf(mx[hh], alpha[e * 4 + hh]);
    }
    #pragma unroll
    for (int hh = 0; hh < 4; hh++) {
        float v = mx[hh];
        for (int off = 16; off > 0; off >>= 1) v = fmaxf(v, __shfl_xor(v, off, 32));
        if (lane == 0) s_w[w][hh] = v;
    }
    __syncthreads();
    if (t < 4)
        s_amax[t] = fmaxf(fmaxf(s_w[0][t], s_w[1][t]), fmaxf(s_w[2][t], s_w[3][t]));
    __syncthreads();

    // pass 2: per-head sum of exp
    float sm[4] = {0.f, 0.f, 0.f, 0.f};
    for (int j = t; j < deg; j += 128) {
        long e = elist[start + j];
        #pragma unroll
        for (int hh = 0; hh < 4; hh++) sm[hh] += __expf(alpha[e * 4 + hh] - s_amax[hh]);
    }
    #pragma unroll
    for (int hh = 0; hh < 4; hh++) {
        float v = sm[hh];
        for (int off = 16; off > 0; off >>= 1) v += __shfl_xor(v, off, 32);
        if (lane == 0) s_w[w][hh] = v;
    }
    __syncthreads();
    if (t < 4)
        s_den[t] = s_w[0][t] + s_w[1][t] + s_w[2][t] + s_w[3][t];
    __syncthreads();

    // pass 3: gather-aggregate messages; thread t owns channel t (head = t>>5)
    const int hh = t >> 5;
    const float wc = wel[t], bc = bel[t];
    float accv = 0.0f;
    for (int base = 0; base < deg; base += 128) {
        int j = base + t;
        if (j < deg) {
            long e = elist[start + j];
            s_src[t] = ei[e];
            s_ea[t]  = eattr[e];
            #pragma unroll
            for (int h2 = 0; h2 < 4; h2++)
                s_a[h2][t] = __expf(alpha[e * 4 + h2] - s_amax[h2]) / s_den[h2];
        }
        __syncthreads();
        const int cn = min(128, deg - base);
        for (int j2 = 0; j2 < cn; j2++) {
            const int src = s_src[j2];
            if (j2 + 1 < cn)
                __builtin_prefetch(qkvs + (long)s_src[j2 + 1] * 512 + 256 + t, 0, 3);
            const float ef = s_ea[j2] * wc + bc;
            accv += (qkvs[(long)src * 512 + 256 + t] + ef) * s_a[hh][j2];
        }
        __syncthreads();
    }
    agg[(long)n * 128 + t] = accv;
}

// ---------------- beta gate + residual + LN ----------------
__global__ void combine_ln(const float* __restrict__ agg, const float* __restrict__ qkvs,
                           const float* __restrict__ wb, const float* __restrict__ g,
                           const float* __restrict__ b, float* __restrict__ h,
                           __bf16* __restrict__ hb) {
    __shared__ float sh[4];
    const int n = blockIdx.x, c = threadIdx.x;
    float out = agg[(long)n * 128 + c];
    float xr  = qkvs[(long)n * 512 + 384 + c];
    float s = out * wb[c] + xr * wb[128 + c] + (out - xr) * wb[256 + c];
    s = blocksum128(s, sh);
    float beta = 1.0f / (1.0f + __expf(-s));
    float val = beta * xr + (1.0f - beta) * out;
    float t = h[(long)n * 128 + c] + val;
    float mean = blocksum128(t, sh) * (1.0f / 128.0f);
    float d = t - mean;
    float var = blocksum128(d * d, sh) * (1.0f / 128.0f);
    float y = d * rsqrtf(var + 1e-5f) * g[c] + b[c];
    h[(long)n * 128 + c]  = y;
    hb[(long)n * 128 + c] = f2bf(y);
}

// ---------------- h = ln(h + f) ----------------
__global__ void addln_kernel(const float* __restrict__ f, const float* __restrict__ g,
                             const float* __restrict__ b, float* __restrict__ h,
                             __bf16* __restrict__ hb) {
    __shared__ float sh[4];
    const int n = blockIdx.x, c = threadIdx.x;
    float t = h[(long)n * 128 + c] + f[(long)n * 128 + c];
    float mean = blocksum128(t, sh) * (1.0f / 128.0f);
    float d = t - mean;
    float var = blocksum128(d * d, sh) * (1.0f / 128.0f);
    float y = d * rsqrtf(var + 1e-5f) * g[c] + b[c];
    h[(long)n * 128 + c]  = y;
    hb[(long)n * 128 + c] = f2bf(y);
}

__global__ void gelu_to_bf16(const float* __restrict__ src, __bf16* __restrict__ dst,
                             long count) {
    long i = (long)blockIdx.x * blockDim.x + threadIdx.x;
    if (i < count) dst[i] = f2bf(gelu_exact(src[i]));
}

// ---------------- pooling ----------------
__global__ void pool_sum(const float* __restrict__ h, const int* __restrict__ batch,
                         float* __restrict__ gsum, float* __restrict__ cnt) {
    const int n = blockIdx.x, c = threadIdx.x;
    const int g = batch[n];
    atomicAdd(&gsum[(long)g * 128 + c], h[(long)n * 128 + c]);
    if (c == 0) atomicAdd(&cnt[g], 1.0f);
}

__global__ void gctx_kernel(const float* __restrict__ gsum, const float* __restrict__ cnt,
                            const float* __restrict__ gw, const float* __restrict__ gb,
                            float* __restrict__ gctx) {
    __shared__ float gm[128];
    const int g = blockIdx.x, c = threadIdx.x;
    float cc = cnt[g];
    cc = cc > 1.0f ? cc : 1.0f;
    gm[c] = gsum[(long)g * 128 + c] / cc;
    __syncthreads();
    float s = gb[c];
    for (int k = 0; k < 128; k++) s += gm[k] * gw[k * 128 + c];
    gctx[(long)g * 128 + c] = s;
}

__global__ void add_ctx(const float* __restrict__ h, const int* __restrict__ batch,
                        const float* __restrict__ gctx, __bf16* __restrict__ hb) {
    const int n = blockIdx.x, c = threadIdx.x;
    float v = h[(long)n * 128 + c] + gctx[(long)batch[n] * 128 + c];
    hb[(long)n * 128 + c] = f2bf(v);
}

__global__ void coords_kernel(const __bf16* __restrict__ t, const float* __restrict__ cw2,
                              const float* __restrict__ cb2, float* __restrict__ out) {
    __shared__ float tr[128];
    const int n = blockIdx.x, c = threadIdx.x;
    tr[c] = bf2f(t[(long)n * 128 + c]);
    __syncthreads();
    if (c < 3) {
        float s = cb2[c];
        for (int k = 0; k < 128; k++) s += tr[k] * cw2[k * 3 + c];
        out[n * 3 + c] = s;
    }
}

// =======================================================================
extern "C" void kernel_launch(void* const* d_in, const int* in_sizes, int n_in,
                              void* d_out, int out_size, void* d_ws, size_t ws_size,
                              hipStream_t stream) {
    const int N = in_sizes[0] / 20;   // 50000
    const int E = in_sizes[1];        // 1600000
    const int L = 4, G = 32;

    const float* x         = (const float*)d_in[0];
    const float* edge_attr = (const float*)d_in[1];
    const int*   edge_idx  = (const int*)  d_in[2];
    const int*   batch     = (const int*)  d_in[3];
    const float* enc_w  = (const float*)d_in[4];
    const float* enc_b  = (const float*)d_in[5];
    const float* enc_g  = (const float*)d_in[6];
    const float* enc_bn = (const float*)d_in[7];
    const float* wq = (const float*)d_in[8];   const float* bq = (const float*)d_in[9];
    const float* wk = (const float*)d_in[10];  const float* bk = (const float*)d_in[11];
    const float* wv = (const float*)d_in[12];  const float* bv = (const float*)d_in[13];
    const float* we = (const float*)d_in[14];  const float* be = (const float*)d_in[15];
    const float* wsk= (const float*)d_in[16];  const float* bsk= (const float*)d_in[17];
    const float* wbeta = (const float*)d_in[18];
    const float* g1 = (const float*)d_in[19];  const float* b1 = (const float*)d_in[20];
    const float* fw1= (const float*)d_in[21];  const float* fb1= (const float*)d_in[22];
    const float* fw2= (const float*)d_in[23];  const float* fb2= (const float*)d_in[24];
    const float* g2 = (const float*)d_in[25];  const float* b2 = (const float*)d_in[26];
    const float* gw = (const float*)d_in[27];  const float* gb = (const float*)d_in[28];
    const float* cw1= (const float*)d_in[29];  const float* cb1= (const float*)d_in[30];
    const float* cw2= (const float*)d_in[31];  const float* cb2= (const float*)d_in[32];
    float* out = (float*)d_out;

    // ---- workspace carve ----
    char* p = (char*)d_ws;
    auto carve = [&](size_t bytes) -> void* {
        void* r = (void*)p;
        p += (bytes + 255) & ~(size_t)255;
        return r;
    };
    float*    h      = (float*)   carve((size_t)N * 128 * 4);
    __bf16*   hb     = (__bf16*)  carve((size_t)N * 128 * 2);
    float*    qkvs   = (float*)   carve((size_t)N * 512 * 4);
    float*    alpha  = (float*)   carve((size_t)E * 4 * 4);
    float*    agg    = (float*)   carve((size_t)N * 128 * 4);
    float*    ffn1   = (float*)   carve((size_t)N * 256 * 4);
    __bf16*   ffn1b  = (__bf16*)  carve((size_t)N * 256 * 2);
    int*      deg    = (int*)     carve((size_t)N * 4);
    int*      rowstart = (int*)   carve((size_t)(N + 1) * 4);
    int*      cursor = (int*)     carve((size_t)N * 4);
    int*      elist  = (int*)     carve((size_t)E * 4);
    __bf16*   wqkvsT = (__bf16*)  carve((size_t)L * 512 * 128 * 2);
    float*    qkvsB  = (float*)   carve((size_t)L * 512 * 4);
    __bf16*   fw1T   = (__bf16*)  carve((size_t)L * 256 * 128 * 2);
    __bf16*   fw2T   = (__bf16*)  carve((size_t)L * 128 * 256 * 2);
    __bf16*   cw1T   = (__bf16*)  carve((size_t)128 * 128 * 2);
    float*    gsum   = (float*)   carve((size_t)G * 128 * 4);
    float*    cnt    = (float*)   carve((size_t)G * 4);
    float*    gctx   = (float*)   carve((size_t)G * 128 * 4);

    // ---- weight conversion (bf16, transposed to out-major) ----
    for (int l = 0; l < L; l++) {
        pack_qkvs<<<(512 * 128 + 255) / 256, 256, 0, stream>>>(
            wq + l * 128 * 128, wk + l * 128 * 128, wv + l * 128 * 128, wsk + l * 128 * 128,
            bq + l * 128, bk + l * 128, bv + l * 128, bsk + l * 128,
            wqkvsT + (size_t)l * 512 * 128, qkvsB + l * 512);
        conv_t_bf16<<<(128 * 256 + 255) / 256, 256, 0, stream>>>(
            fw1 + (size_t)l * 128 * 256, fw1T + (size_t)l * 256 * 128, 128, 256);
        conv_t_bf16<<<(256 * 128 + 255) / 256, 256, 0, stream>>>(
            fw2 + (size_t)l * 256 * 128, fw2T + (size_t)l * 128 * 256, 256, 128);
    }
    conv_t_bf16<<<(128 * 128 + 255) / 256, 256, 0, stream>>>(cw1, cw1T, 128, 128);

    // ---- CSR build over dst (once; edge_index constant across layers) ----
    hipMemsetAsync(deg,    0, (size_t)N * 4, stream);
    hipMemsetAsync(cursor, 0, (size_t)N * 4, stream);
    deg_kernel<<<(E + 255) / 256, 256, 0, stream>>>(edge_idx, deg, E);
    scan_kernel<<<1, 1024, 0, stream>>>(deg, rowstart, N);
    fill_kernel<<<(E + 255) / 256, 256, 0, stream>>>(edge_idx, rowstart, cursor, elist, E);

    // ---- encoder ----
    encoder_kernel<<<N, 128, 0, stream>>>(x, enc_w, enc_b, enc_g, enc_bn, h, hb);

    const int rowTiles = N / 16;               // 50000 = 16*3125
    const int edgeBlocks = (E + 7) / 8;

    for (int l = 0; l < L; l++) {
        const float* wel = we + l * 128;
        const float* bel = be + l * 128;

        // fused q|k|v|skip projection: [N x 128] @ [128 x 512]  (8 waves/block)
        wmma_gemm_bf16<128><<<rowTiles, 32 * (512 / 64), 0, stream>>>(
            hb, wqkvsT + (size_t)l * 512 * 128, qkvsB + l * 512, qkvs, 512);

        edge_alpha<<<edgeBlocks, 256, 0, stream>>>(edge_idx, edge_attr, qkvs, wel, bel,
                                                   alpha, E);
        node_softmax_agg<<<N, 128, 0, stream>>>(edge_idx, edge_attr, rowstart, elist,
                                                qkvs, wel, bel, alpha, agg, E);

        combine_ln<<<N, 128, 0, stream>>>(agg, qkvs, wbeta + l * 384,
                                          g1 + l * 128, b1 + l * 128, h, hb);

        // FFN
        wmma_gemm_bf16<128><<<rowTiles, 32 * (256 / 64), 0, stream>>>(
            hb, fw1T + (size_t)l * 256 * 128, fb1 + l * 256, ffn1, 256);
        gelu_to_bf16<<<((long)N * 256 + 255) / 256, 256, 0, stream>>>(ffn1, ffn1b, (long)N * 256);
        wmma_gemm_bf16<256><<<rowTiles, 32 * (128 / 64), 0, stream>>>(
            ffn1b, fw2T + (size_t)l * 128 * 256, fb2 + l * 128, agg, 128);
        addln_kernel<<<N, 128, 0, stream>>>(agg, g2 + l * 128, b2 + l * 128, h, hb);
    }

    // ---- global pooling + context ----
    hipMemsetAsync(gsum, 0, (size_t)G * 128 * 4, stream);
    hipMemsetAsync(cnt,  0, (size_t)G * 4, stream);
    pool_sum<<<N, 128, 0, stream>>>(h, batch, gsum, cnt);
    gctx_kernel<<<G, 128, 0, stream>>>(gsum, cnt, gw, gb, gctx);
    add_ctx<<<N, 128, 0, stream>>>(h, batch, gctx, hb);

    // ---- decoder: gelu(h2 @ cw1 + cb1) @ cw2 + cb2 ----
    wmma_gemm_bf16<128><<<rowTiles, 32 * (128 / 64), 0, stream>>>(hb, cw1T, cb1, ffn1, 128);
    gelu_to_bf16<<<((long)N * 128 + 255) / 256, 256, 0, stream>>>(ffn1, ffn1b, (long)N * 128);
    coords_kernel<<<N, 128, 0, stream>>>(ffn1b, cw2, cb2, out);
}
